// CrossAttention_8486855377097
// MI455X (gfx1250) — matrix operand order repaired and checked by reading, not compiled, not run on hardware
//
#include <hip/hip_runtime.h>
#include <hip/hip_bf16.h>

// ---------------- problem constants ----------------
#define BATCH 2
#define NQ    416
#define NCTX  257
#define DMODEL 768
#define NHEAD 6
#define DHEAD 128
#define NGRP  104
#define RANK  64
#define GR    (NGRP * RANK)     // 6656
#define LN_EPS 1e-5f

typedef __bf16 bf16_t;
typedef __attribute__((ext_vector_type(16))) __bf16 v16bf;
typedef __attribute__((ext_vector_type(8)))  __bf16 v8bf;
typedef __attribute__((ext_vector_type(8)))  float  v8f;

static __device__ __forceinline__ v16bf cat16(v8bf lo, v8bf hi) {
  return __builtin_shufflevector(lo, hi, 0, 1, 2, 3, 4, 5, 6, 7,
                                 8, 9, 10, 11, 12, 13, 14, 15);
}
static __device__ __forceinline__ v8bf ldg8(const bf16_t* p) {
  return *reinterpret_cast<const v8bf*>(p);
}

// ---- CDNA5 async global->LDS copy (tracked by ASYNCcnt) --------------------
static __device__ __forceinline__ void async_copy_b128(void* lds_dst,
                                                       const void* gsrc) {
  unsigned lds_off = (unsigned)(uintptr_t)lds_dst;  // LDS aperture low bits
  asm volatile("global_load_async_to_lds_b128 %0, %1, off"
               :: "v"(lds_off), "v"(gsrc)
               : "memory");
}
static __device__ __forceinline__ void wait_async0() {
  asm volatile("s_wait_asynccnt 0x0" ::: "memory");
}

// ---------------------------------------------------------------------------
// Elementwise f32 -> bf16 convert
// ---------------------------------------------------------------------------
__global__ void cvt_f32_to_bf16(const float* __restrict__ in,
                                bf16_t* __restrict__ out, int n) {
  int i = blockIdx.x * blockDim.x + threadIdx.x;
  if (i < n) out[i] = (bf16_t)in[i];
}

// W (K, N) f32 -> Wt (N, K) bf16   (coalesced reads; one-time cost)
__global__ void transpose_cvt_bf16(const float* __restrict__ in,
                                   bf16_t* __restrict__ out, int K, int N) {
  int i = blockIdx.x * blockDim.x + threadIdx.x;
  if (i >= K * N) return;
  int k = i / N, n = i % N;
  out[(size_t)n * K + k] = (bf16_t)in[i];
}

// ---------------------------------------------------------------------------
// Generic bf16 GEMM: C(MxN, f32) = A(MxK, bf16) * Bt(NxK, bf16  == B^T)
// One wave32 computes a 16x64 strip: A fragment reused across 4 N-tiles.
// All fragment loads are 128-bit. A rows beyond M read padded workspace.
// ---------------------------------------------------------------------------
#define GNT 4
__global__ __launch_bounds__(32) void gemm_bf16_wmma(
    const bf16_t* __restrict__ A, const bf16_t* __restrict__ Bt,
    float* __restrict__ C, int M, int N, int K) {
  const int tn = blockIdx.x;            // N / 64 strips
  const int tm = blockIdx.y;            // ceil(M / 16) tiles (A padded)
  const int lane = threadIdx.x;
  const int lr = lane & 15;
  const int hi = lane >> 4;

  const bf16_t* arow = A + (size_t)(tm * 16 + lr) * K;

  v8f acc[GNT];
#pragma unroll
  for (int t = 0; t < GNT; ++t) acc[t] = v8f{};

  for (int k0 = 0; k0 < K; k0 += 32) {
    __builtin_prefetch(arow + k0 + 128, 0, 1);
    v16bf a = cat16(ldg8(arow + k0 + hi * 16), ldg8(arow + k0 + hi * 16 + 8));
#pragma unroll
    for (int t = 0; t < GNT; ++t) {
      const bf16_t* brow =
          Bt + (size_t)(tn * 64 + t * 16 + lr) * K + k0 + hi * 16;
      v16bf b = cat16(ldg8(brow), ldg8(brow + 8));
      acc[t] = __builtin_amdgcn_wmma_f32_16x16x32_bf16(false, a, false, b,
                                                       (short)0, acc[t],
                                                       false, false);
    }
  }
#pragma unroll
  for (int t = 0; t < GNT; ++t) {
#pragma unroll
    for (int r = 0; r < 8; ++r) {
      int mo = tm * 16 + r + 8 * hi;
      if (mo < M) C[(size_t)mo * N + tn * 64 + t * 16 + lr] = acc[t][r];
    }
  }
}

// ---------------------------------------------------------------------------
// Row LayerNorm: one block (256 thr) per row. Optional f32 and/or bf16 output.
// Safe for in-place (out_f == x).
// ---------------------------------------------------------------------------
__global__ __launch_bounds__(256) void layernorm_rows(
    const float* x, const float* __restrict__ gam,
    const float* __restrict__ bet, int C, float* out_f, bf16_t* out_bf) {
  const int row = blockIdx.x;
  const int tid = threadIdx.x;
  const float* xr = x + (size_t)row * C;
  __shared__ float red[256], red2[256];
  __shared__ float sm, sr;

  float s = 0.f, s2 = 0.f;
  for (int c = tid; c < C; c += 256) { float v = xr[c]; s += v; s2 += v * v; }
  red[tid] = s; red2[tid] = s2;
  __syncthreads();
  for (int off = 128; off > 0; off >>= 1) {
    if (tid < off) { red[tid] += red[tid + off]; red2[tid] += red2[tid + off]; }
    __syncthreads();
  }
  if (tid == 0) {
    float mean = red[0] / (float)C;
    float var  = red2[0] / (float)C - mean * mean;
    sm = mean; sr = rsqrtf(var + LN_EPS);
  }
  __syncthreads();
  const float mean = sm, rstd = sr;
  for (int c = tid; c < C; c += 256) {
    float y = (xr[c] - mean) * rstd * gam[c] + bet[c];
    if (out_f)  out_f[(size_t)row * C + c]  = y;
    if (out_bf) out_bf[(size_t)row * C + c] = (bf16_t)y;
  }
}

// ---------------------------------------------------------------------------
// Fused q-aware value generation:
//   vraw[b,g,n,:] = mid[b,n,g,:] @ conv_w[g]^T    (257x64 @ 64x768, WMMA)
//   v = LN(vraw) * v_g + v_b + context[b,n,:]     (fused in LDS)
// conv_w (G, D, RANK) bf16 is already the Bt (N, K) layout per group.
// Block: 256 thr = 8 waves; each wave owns 6 N-tiles; 16-row tile over NCTX.
// ---------------------------------------------------------------------------
__global__ __launch_bounds__(256) void vgen_wmma(
    const bf16_t* __restrict__ midbf,  // (B, NCTX, GR), row-padded
    const bf16_t* __restrict__ convbf, // (G, D, RANK)
    const float* __restrict__ ctx,     // (B, NCTX, D)
    const float* __restrict__ gam, const float* __restrict__ bet,
    bf16_t* __restrict__ vout) {       // (B, G, NCTX, D)
  const int tile = blockIdx.x;         // 17 tiles of 16 rows over NCTX
  const int g    = blockIdx.y;
  const int b    = blockIdx.z;
  const int tid  = threadIdx.x;
  const int wave = tid >> 5, lane = tid & 31;
  const int lr   = lane & 15, hi = lane >> 4;

  __shared__ float vtile[16][DMODEL];  // 48 KB
  __shared__ float rsum[16][16], rsum2[16][16];
  __shared__ float smean[16], srstd[16];

  v8f acc[6];
#pragma unroll
  for (int t = 0; t < 6; ++t) acc[t] = v8f{};

  const bf16_t* arow =
      midbf + (size_t)(b * NCTX + tile * 16 + lr) * GR + g * RANK;
#pragma unroll
  for (int k0 = 0; k0 < RANK; k0 += 32) {
    v16bf a = cat16(ldg8(arow + k0 + hi * 16), ldg8(arow + k0 + hi * 16 + 8));
#pragma unroll
    for (int t = 0; t < 6; ++t) {
      const int n0 = (wave * 6 + t) * 16;
      const bf16_t* brow =
          convbf + ((size_t)g * DMODEL + n0 + lr) * RANK + k0 + hi * 16;
      v16bf bfr = cat16(ldg8(brow), ldg8(brow + 8));
      acc[t] = __builtin_amdgcn_wmma_f32_16x16x32_bf16(false, a, false, bfr,
                                                       (short)0, acc[t],
                                                       false, false);
    }
  }

  // spill C fragments to LDS for the row-wise LN
#pragma unroll
  for (int t = 0; t < 6; ++t) {
    const int n0 = (wave * 6 + t) * 16;
#pragma unroll
    for (int r = 0; r < 8; ++r) vtile[r + 8 * hi][n0 + lr] = acc[t][r];
  }
  __syncthreads();

  // LN over D=768 per row: 16 threads per row
  const int r  = tid >> 4;
  const int cs = tid & 15;
  float s = 0.f, s2 = 0.f;
  for (int c = cs; c < DMODEL; c += 16) {
    float v = vtile[r][c]; s += v; s2 += v * v;
  }
  rsum[r][cs] = s; rsum2[r][cs] = s2;
  __syncthreads();
  if (cs == 0) {
    float ts = 0.f, ts2 = 0.f;
#pragma unroll
    for (int i = 0; i < 16; ++i) { ts += rsum[r][i]; ts2 += rsum2[r][i]; }
    float mean = ts / (float)DMODEL;
    float var  = ts2 / (float)DMODEL - mean * mean;
    smean[r] = mean; srstd[r] = rsqrtf(var + LN_EPS);
  }
  __syncthreads();
  const int growr = tile * 16 + r;
  if (growr < NCTX) {
    const float mean = smean[r], rstd = srstd[r];
    for (int c = cs; c < DMODEL; c += 16) {
      float y = (vtile[r][c] - mean) * rstd * gam[c] + bet[c]
              + ctx[((size_t)b * NCTX + growr) * DMODEL + c];
      vout[(((size_t)b * NGRP + g) * NCTX + growr) * DMODEL + c] = (bf16_t)y;
    }
  }
}

// ---------------------------------------------------------------------------
// Attention: one 256-thr block per (b, h, query). NCTX=257 is WMMA-hostile
// and <5% of total FLOPs; v is bf16 and L2-resident. The value contraction
// stages v in 16-row chunks into LDS via CDNA5 async global->LDS copies
// (ASYNCcnt), hiding L2 latency behind the accumulate loop.
// ---------------------------------------------------------------------------
#define VCH 16
__global__ __launch_bounds__(256) void attention_kernel(
    const float* __restrict__ q,   // (B, NQ, D)
    const float* __restrict__ k,   // (B, NCTX, D)
    const bf16_t* __restrict__ v,  // (B, G, NCTX, D)
    bf16_t* __restrict__ out) {    // (B, NQ, D)
  const int i = blockIdx.x, h = blockIdx.y, b = blockIdx.z;
  const int tid = threadIdx.x;
  __shared__ __align__(16) float qs[DHEAD];
  __shared__ __align__(16) bf16_t vstage[VCH][DHEAD];  // 4 KB chunk
  __shared__ float p[NCTX];
  __shared__ float red[256];
  __shared__ float sacc[256];
  __shared__ float smax, ssum;

  if (tid < DHEAD) qs[tid] = q[((size_t)b * NQ + i) * DMODEL + h * DHEAD + tid];
  __syncthreads();

  const float scale = 0.088388347648318447f;  // DH^-0.5
  const float4* qs4 = reinterpret_cast<const float4*>(qs);
  float mymax = -3.0e38f;
  for (int j = tid; j < NCTX; j += 256) {
    const float4* kr4 = reinterpret_cast<const float4*>(
        k + ((size_t)b * NCTX + j) * DMODEL + h * DHEAD);
    float dot = 0.f;
#pragma unroll
    for (int d4 = 0; d4 < DHEAD / 4; ++d4) {
      float4 a = qs4[d4], bb = kr4[d4];
      dot += a.x * bb.x + a.y * bb.y + a.z * bb.z + a.w * bb.w;
    }
    dot *= scale;
    p[j] = dot;
    mymax = fmaxf(mymax, dot);
  }
  red[tid] = mymax;
  __syncthreads();
  for (int off = 128; off > 0; off >>= 1) {
    if (tid < off) red[tid] = fmaxf(red[tid], red[tid + off]);
    __syncthreads();
  }
  if (tid == 0) smax = red[0];
  __syncthreads();
  const float mx = smax;
  float mysum = 0.f;
  for (int j = tid; j < NCTX; j += 256) {
    float e = __expf(p[j] - mx);
    p[j] = e;
    mysum += e;
  }
  red[tid] = mysum;
  __syncthreads();
  for (int off = 128; off > 0; off >>= 1) {
    if (tid < off) red[tid] += red[tid + off];
    __syncthreads();
  }
  if (tid == 0) ssum = red[0];
  __syncthreads();
  const float inv = 1.0f / ssum;

  const int g = i % NGRP;               // jnp.tile along q-group axis
  const int d = tid & (DHEAD - 1);
  const int sseg = tid >> 7;            // split j between thread halves
  const bf16_t* vbase =
      v + (((size_t)b * NGRP + g) * NCTX) * DMODEL + h * DHEAD;

  float acc = 0.f;
  const int srow = tid >> 4;            // staging: 16 rows x 16 segments
  const int sseg16 = tid & 15;          // 16B (8 x bf16) segment per thread
  for (int j0 = 0; j0 < NCTX; j0 += VCH) {
    const int rows = (NCTX - j0 < VCH) ? (NCTX - j0) : VCH;
    if (srow < rows)
      async_copy_b128(&vstage[srow][sseg16 * 8],
                      vbase + (size_t)(j0 + srow) * DMODEL + sseg16 * 8);
    wait_async0();
    __syncthreads();
    for (int jj = sseg; jj < rows; jj += 2)
      acc += p[j0 + jj] * (float)vstage[jj][d];
    __syncthreads();
  }
  sacc[tid] = acc;
  __syncthreads();
  if (tid < DHEAD) {
    float o = (sacc[tid] + sacc[tid + 128]) * inv;
    out[((size_t)b * NQ + i) * DMODEL + h * DHEAD + tid] = (bf16_t)o;
  }
}

// ---------------------------------------------------------------------------
// Host-side orchestration
// ---------------------------------------------------------------------------
extern "C" void kernel_launch(void* const* d_in, const int* in_sizes, int n_in,
                              void* d_out, int out_size, void* d_ws,
                              size_t ws_size, hipStream_t stream) {
  (void)in_sizes; (void)n_in; (void)out_size; (void)ws_size;

  const float* x     = (const float*)d_in[0];
  const float* ctx   = (const float*)d_in[1];
  const float* Wq    = (const float*)d_in[2];
  const float* q_g   = (const float*)d_in[3];
  const float* q_b   = (const float*)d_in[4];
  const float* Wk    = (const float*)d_in[5];
  const float* k_g   = (const float*)d_in[6];
  const float* k_b   = (const float*)d_in[7];
  const float* Wv    = (const float*)d_in[8];
  const float* mid_g = (const float*)d_in[9];
  const float* mid_b = (const float*)d_in[10];
  const float* convw = (const float*)d_in[11];
  const float* v_g   = (const float*)d_in[12];
  const float* v_b   = (const float*)d_in[13];
  const float* Wo    = (const float*)d_in[14];
  const float* o_g   = (const float*)d_in[15];
  const float* o_b   = (const float*)d_in[16];
  float* out = (float*)d_out;

  // ---- workspace carving (A-side buffers padded to whole 16-row tiles) ----
  char* ws = (char*)d_ws;
  size_t off = 0;
  auto carve = [&](size_t bytes) -> void* {
    void* p = ws + off;
    off = (off + bytes + 255) & ~(size_t)255;
    return p;
  };
  const int Mq = BATCH * NQ;            // 832  (52 exact tiles)
  const int Mc = BATCH * NCTX;          // 514  -> 33 tiles -> 528 rows read
  const int tMq = Mq / 16, tMc = (Mc + 15) / 16;

  const size_t nX      = (size_t)BATCH * NQ * DMODEL;        // 638976
  const size_t nCtx    = (size_t)BATCH * NCTX * DMODEL;      // 394752
  const size_t nCtxPad = (size_t)(tMc * 16) * DMODEL;        // 528 rows
  const size_t nW      = (size_t)DMODEL * DMODEL;
  const size_t nWv     = (size_t)DMODEL * GR;
  const size_t nCw     = (size_t)NGRP * DMODEL * RANK;
  const size_t nMid    = (size_t)Mc * GR;
  const size_t nMidPad = (size_t)(NCTX + 17 * 16) * GR;      // b*NCTX+row<=528
  const size_t nV      = (size_t)BATCH * NGRP * NCTX * DMODEL;

  bf16_t* xbf    = (bf16_t*)carve(nX      * sizeof(bf16_t));
  bf16_t* cbf    = (bf16_t*)carve(nCtxPad * sizeof(bf16_t));  // padded rows
  bf16_t* wqT    = (bf16_t*)carve(nW      * sizeof(bf16_t));  // (N,K)
  bf16_t* wkT    = (bf16_t*)carve(nW      * sizeof(bf16_t));
  bf16_t* woT    = (bf16_t*)carve(nW      * sizeof(bf16_t));
  bf16_t* wvT    = (bf16_t*)carve(nWv     * sizeof(bf16_t));  // (GR, D)
  bf16_t* convbf = (bf16_t*)carve(nCw     * sizeof(bf16_t));  // (G, D, RANK)
  float*  qf     = (float*) carve(nX      * sizeof(float));
  float*  kf     = (float*) carve(nCtx    * sizeof(float));
  float*  midf   = (float*) carve(nMid    * sizeof(float));
  bf16_t* midbf  = (bf16_t*)carve(nMidPad * sizeof(bf16_t));  // padded rows
  bf16_t* vbf    = (bf16_t*)carve(nV      * sizeof(bf16_t));
  bf16_t* aobf   = (bf16_t*)carve(nX      * sizeof(bf16_t));
  float*  olin   = (float*) carve(nX      * sizeof(float));

  auto eb = [](size_t n) { return dim3((unsigned)((n + 255) / 256)); };

  // 1) convert / transpose weights and activations to bf16
  cvt_f32_to_bf16<<<eb(nX),   256, 0, stream>>>(x,   xbf,  (int)nX);
  cvt_f32_to_bf16<<<eb(nCtx), 256, 0, stream>>>(ctx, cbf,  (int)nCtx);
  cvt_f32_to_bf16<<<eb(nCw),  256, 0, stream>>>(convw, convbf, (int)nCw);
  transpose_cvt_bf16<<<eb(nW),  256, 0, stream>>>(Wq, wqT, DMODEL, DMODEL);
  transpose_cvt_bf16<<<eb(nW),  256, 0, stream>>>(Wk, wkT, DMODEL, DMODEL);
  transpose_cvt_bf16<<<eb(nW),  256, 0, stream>>>(Wo, woT, DMODEL, DMODEL);
  transpose_cvt_bf16<<<eb(nWv), 256, 0, stream>>>(Wv, wvT, DMODEL, GR);

  // 2) projections (WMMA GEMMs; B supplied transposed)
  gemm_bf16_wmma<<<dim3(DMODEL / 64, tMq), 32, 0, stream>>>(
      xbf, wqT, qf, Mq, DMODEL, DMODEL);
  gemm_bf16_wmma<<<dim3(DMODEL / 64, tMc), 32, 0, stream>>>(
      cbf, wkT, kf, Mc, DMODEL, DMODEL);
  gemm_bf16_wmma<<<dim3(GR / 64, tMc), 32, 0, stream>>>(
      cbf, wvT, midf, Mc, GR, DMODEL);

  // 3) LayerNorms (q,k in-place f32; mid -> bf16 for the grouped conv)
  layernorm_rows<<<Mq, 256, 0, stream>>>(qf, q_g, q_b, DMODEL, qf, nullptr);
  layernorm_rows<<<Mc, 256, 0, stream>>>(kf, k_g, k_b, DMODEL, kf, nullptr);
  layernorm_rows<<<Mc, 256, 0, stream>>>(midf, mid_g, mid_b, GR, nullptr, midbf);

  // 4) fused grouped-conv GEMM + LN + skip -> v (bf16, L2-resident)
  vgen_wmma<<<dim3((NCTX + 15) / 16, NGRP, BATCH), 256, 0, stream>>>(
      midbf, convbf, ctx, v_g, v_b, vbf);

  // 5) attention with per-query value groups (async LDS staging of v)
  attention_kernel<<<dim3(NQ, NHEAD, BATCH), 256, 0, stream>>>(qf, kf, vbf,
                                                               aobf);

  // 6) output projection + final LN into d_out
  gemm_bf16_wmma<<<dim3(DMODEL / 64, tMq), 32, 0, stream>>>(
      aobf, woT, olin, Mq, DMODEL, DMODEL);
  layernorm_rows<<<Mq, 256, 0, stream>>>(olin, o_g, o_b, DMODEL, out, nullptr);
}
